// TemporalTransformer_58093727645829
// MI455X (gfx1250) — matrix operand order repaired and checked by reading, not compiled, not run on hardware
//
#include <hip/hip_runtime.h>
#include <math.h>

// ---------------- problem constants ----------------
namespace {
constexpr int cB = 2, cT = 16, cT1 = 17, cH = 16, cW = 16, cHW = 256;
constexpr int cD = 512, cNH = 8, cDH = 64, cDEPTH = 6, cMLP = 1024;
constexpr float NEGV = -1e9f;
constexpr int ROWS_SP = cB * cT * cHW;        // 8192 spatial tokens
constexpr int ROWS_T  = cB * cHW * cT1;       // 8704 temporal tokens
}

typedef __bf16 bf16;
typedef __attribute__((ext_vector_type(16))) __bf16 v16bf;
typedef __attribute__((ext_vector_type(8)))  __bf16 v8bf;
typedef __attribute__((ext_vector_type(8)))  float  v8f;

// Build a WMMA fragment from two contiguous 16B loads.
// A and B fragments: hi half at +16 elements.
__device__ __forceinline__ v16bf mk_frag(const bf16* p, int hi_off) {
  v8bf lo = *(const v8bf*)p;
  v8bf hi = *(const v8bf*)(p + hi_off);
  v16bf r;
#pragma unroll
  for (int i = 0; i < 8; ++i) { r[i] = lo[i]; r[i + 8] = hi[i]; }
  return r;
}

// =====================================================================
// bf16 WMMA GEMM:  C[M,N] = act( A[M,K](bf16) @ W^T[N,K](bf16) + bias[N] (+resid) )
// Wt is the K-major (transposed) weight.  Optional output-row remap:
// phys_row = r + (r/chunk)*gap  (chunk==0 -> identity).  resid may alias C.
// Block = 8 waves; wave computes 32(M) x 64(N); block tile 256 x 64.
// K-loop is software-pipelined with ping-pong register fragment buffers so
// global loads for chunk k+32 are in flight while WMMAs for chunk k execute.
// =====================================================================
__global__ __launch_bounds__(256) void gemm_wmma(
    const bf16* __restrict__ A, const bf16* __restrict__ Wt,
    const float* __restrict__ bias, const float* resid, float* C,
    int M, int N, int K, int do_tanh, int chunk, int gap)
{
  const int wave = threadIdx.x >> 5;
  const int lane = threadIdx.x & 31;
  const int m0 = blockIdx.x * 256 + wave * 32;
  const int n0 = blockIdx.y * 64;
  if (m0 >= M) return;

  const int arow = lane & 15;
  const int asel = (lane >> 4) << 3;   // A: lanes>=16 hold K+8 group
  const int bcol = lane & 15;
  const int bsel = (lane >> 4) << 3;   // B: lanes>=16 hold K+8 group

  const bf16* ApA = A + (size_t)(m0 + arow) * K + asel;        // rows m0..m0+15
  const bf16* ApB = ApA + (size_t)16 * K;                      // rows m0+16..m0+31
  const bf16* Wp  = Wt + (size_t)(n0 + bcol) * K + bsel;       // + t*16*K per frag

  v8f acc[2][4];
#pragma unroll
  for (int p = 0; p < 2; ++p)
#pragma unroll
    for (int t = 0; t < 4; ++t)
#pragma unroll
      for (int r = 0; r < 8; ++r) acc[p][t][r] = 0.f;

  v16bf A0[2], A1[2], Bf[2][4];

#define LOADF(buf, kk)                                                        \
  do {                                                                        \
    A0[buf] = mk_frag(ApA + (kk), 16);                                        \
    A1[buf] = mk_frag(ApB + (kk), 16);                                        \
    _Pragma("unroll")                                                         \
    for (int t = 0; t < 4; ++t)                                               \
      Bf[buf][t] = mk_frag(Wp + (size_t)t * 16 * K + (kk), 16);               \
  } while (0)

#define MMAC(buf)                                                             \
  do {                                                                        \
    _Pragma("unroll")                                                         \
    for (int t = 0; t < 4; ++t) {                                             \
      acc[0][t] = __builtin_amdgcn_wmma_f32_16x16x32_bf16(                    \
          false, A0[buf], false, Bf[buf][t], (short)0, acc[0][t], false, false);\
      acc[1][t] = __builtin_amdgcn_wmma_f32_16x16x32_bf16(                    \
          false, A1[buf], false, Bf[buf][t], (short)0, acc[1][t], false, false);\
    }                                                                         \
  } while (0)

  // K is a multiple of 64 (512 or 1024): 2 chunks of 32 per loop iteration.
  LOADF(0, 0);
  int k0 = 0;
  for (; k0 < K - 64; k0 += 64) {
    LOADF(1, k0 + 32);   // prefetch chunk k0+32
    MMAC(0);             // consume chunk k0
    LOADF(0, k0 + 64);   // prefetch chunk k0+64
    MMAC(1);             // consume chunk k0+32
  }
  LOADF(1, K - 32);
  MMAC(0);
  MMAC(1);
#undef LOADF
#undef MMAC

#pragma unroll
  for (int p = 0; p < 2; ++p) {
    const int mb = m0 + p * 16 + ((lane >> 4) << 3);
#pragma unroll
    for (int t = 0; t < 4; ++t) {
      const int ncol = n0 + t * 16 + (lane & 15);
      const float bb = bias[ncol];
#pragma unroll
      for (int r = 0; r < 8; ++r) {
        int rm = mb + r;
        int pr = chunk ? rm + (rm / chunk) * gap : rm;
        size_t off = (size_t)pr * N + ncol;
        float v = acc[p][t][r] + bb;
        if (resid) v += resid[off];
        if (do_tanh) v = tanhf(v);
        C[off] = v;
      }
    }
  }
}

// ---------------- weight convert + transpose: [depth,K,N] f32 -> [depth,N,K] bf16
__global__ __launch_bounds__(256) void wcvt(const float* __restrict__ in,
                                            bf16* __restrict__ out,
                                            int K, int N, int total)
{
  int idx = blockIdx.x * 256 + threadIdx.x;
  if (idx >= total) return;
  int kn = K * N;
  int layer = idx / kn;
  int rem = idx - layer * kn;
  int k = rem / N, n = rem - k * N;
  out[(size_t)layer * kn + (size_t)n * K + k] = (bf16)in[idx];
}

// ---------------- f32 -> bf16 elementwise
__global__ __launch_bounds__(256) void cvt_bf16(const float* __restrict__ in,
                                                bf16* __restrict__ out, int n)
{
  int i = blockIdx.x * 256 + threadIdx.x;
  if (i < n) out[i] = (bf16)in[i];
}

// ---------------- LayerNorm over D=512, one block per row; optional src row remap
template <typename OutT>
__global__ __launch_bounds__(256) void ln_k(const float* __restrict__ x,
                                            const float* __restrict__ g,
                                            const float* __restrict__ bta,
                                            OutT* __restrict__ out,
                                            int chunk, int gap)
{
  int r = blockIdx.x;
  int src = chunk ? r + (r / chunk) * gap : r;
  int tid = threadIdx.x;
  const float* xr = x + (size_t)src * cD;
  float v0 = xr[tid], v1 = xr[tid + 256];
  __shared__ float red[256];
  red[tid] = v0 + v1; __syncthreads();
  for (int o = 128; o; o >>= 1) { if (tid < o) red[tid] += red[tid + o]; __syncthreads(); }
  float mean = red[0] * (1.f / cD);
  __syncthreads();
  float d0 = v0 - mean, d1 = v1 - mean;
  red[tid] = d0 * d0 + d1 * d1; __syncthreads();
  for (int o = 128; o; o >>= 1) { if (tid < o) red[tid] += red[tid + o]; __syncthreads(); }
  float rs = rsqrtf(red[0] * (1.f / cD) + 1e-5f);
  OutT* orow = out + (size_t)r * cD;
  orow[tid]       = (OutT)(d0 * rs * g[tid] + bta[tid]);
  orow[tid + 256] = (OutT)(d1 * rs * g[tid + 256] + bta[tid + 256]);
}

// ---------------- embedding + positional ----------------
__global__ __launch_bounds__(256) void embed_k(const int* __restrict__ codes,
    const float* __restrict__ emb, const float* __restrict__ cls,
    const float* __restrict__ he, const float* __restrict__ we,
    float* __restrict__ X)
{
  int row = blockIdx.x;                       // b*T1*HW + t*HW + hw
  int b = row / (cT1 * cHW);
  int r2 = row - b * (cT1 * cHW);
  int t = r2 >> 8, hw = r2 & 255;
  for (int d = threadIdx.x; d < cD; d += 256) {
    float v;
    if (t < cT) {
      int c = codes[((b * cT + t) * cHW) + hw];
      v = emb[(size_t)c * cD + d] + he[(hw & 15) * cD + d] + we[(hw >> 4) * cD + d];
    } else {
      v = cls[d];
    }
    X[(size_t)row * cD + d] = v;
  }
}

// ---------------- transpose X[B,T1,HW,D] <-> S[B*HW,T1,D] ----------------
__global__ __launch_bounds__(256) void xpose(const float* __restrict__ src,
                                             float* __restrict__ dst, int x_to_s)
{
  int row = blockIdx.x;                       // enumerates X-layout row
  int b = row / (cT1 * cHW);
  int r2 = row - b * (cT1 * cHW);
  int t = r2 >> 8, hw = r2 & 255;
  size_t xoff = (size_t)row * cD;
  size_t soff = ((size_t)((b * cHW + hw) * cT1 + t)) * cD;
  size_t so = x_to_s ? xoff : soff;
  size_t dd = x_to_s ? soff : xoff;
  dst[dd + threadIdx.x]       = src[so + threadIdx.x];
  dst[dd + threadIdx.x + 256] = src[so + threadIdx.x + 256];
}

__device__ __forceinline__ float wave_sum(float p) {
#pragma unroll
  for (int m = 16; m >= 1; m >>= 1) p += __shfl_xor(p, m, 32);
  return p;
}

// ---------------- spatial 3x3 windowed attention: one wave per (query,head) ----
__global__ __launch_bounds__(256) void spatial_attn(
    const float* __restrict__ Q, const float* __restrict__ K,
    const float* __restrict__ V, float* __restrict__ O)
{
  int wid = blockIdx.x * 8 + (threadIdx.x >> 5);
  int lane = threadIdx.x & 31;
  int head = wid & 7;
  int rq = wid >> 3;                          // 0..8191
  if (rq >= ROWS_SP) return;
  int b = rq >> 12, t = (rq >> 8) & 15, h = (rq >> 4) & 15, w = rq & 15;
  int col = head * cDH + lane;
  float q0 = Q[(size_t)rq * cD + col] * 0.125f;
  float q1 = Q[(size_t)rq * cD + col + 32] * 0.125f;
  float lg[9];
#pragma unroll
  for (int s = 0; s < 9; ++s) {
    int hh = h + s / 3 - 1, ww = w + s % 3 - 1;
    float d = NEGV;
    if (hh >= 0 && hh < cH && ww >= 0 && ww < cW) {
      size_t rk = ((size_t)((b * cT + t) * cHW + hh * cW + ww)) * cD + col;
      d = wave_sum(q0 * K[rk] + q1 * K[rk + 32]);
    }
    lg[s] = d;
  }
  float mx = lg[0];
#pragma unroll
  for (int s = 1; s < 9; ++s) mx = fmaxf(mx, lg[s]);
  float p[9], sum = 0.f;
#pragma unroll
  for (int s = 0; s < 9; ++s) { p[s] = expf(lg[s] - mx); sum += p[s]; }
  float inv = 1.f / sum;
  float o0 = 0.f, o1 = 0.f;
#pragma unroll
  for (int s = 0; s < 9; ++s) {
    int hh = h + s / 3 - 1, ww = w + s % 3 - 1;
    if (hh >= 0 && hh < cH && ww >= 0 && ww < cW) {
      size_t rk = ((size_t)((b * cT + t) * cHW + hh * cW + ww)) * cD + col;
      o0 += p[s] * inv * V[rk];
      o1 += p[s] * inv * V[rk + 32];
    }
  }
  O[(size_t)rq * cD + col] = o0;
  O[(size_t)rq * cD + col + 32] = o1;
}

// ---------------- RoPE on temporal Q,K (in place) + export k/v caches --------
__global__ __launch_bounds__(256) void rope_export(
    float* Q, float* K, const float* __restrict__ V,
    float* __restrict__ kout, float* __restrict__ vout)
{
  int row = blockIdx.x;                       // 0..8703 = n*17+t
  int n = row / cT1, t = row - n * cT1;
  int tid = threadIdx.x;
  float qn[2], kn[2];
  int ds[2] = { tid, tid + 256 };
#pragma unroll
  for (int e = 0; e < 2; ++e) {
    int d = ds[e];
    int head = d >> 6, dd = d & 63, j = dd & 31;
    float ang = (float)t * powf(10000.f, -(float)j / 32.f);
    float c = cosf(ang), s = sinf(ang);
    size_t off = (size_t)row * cD + d;
    int pd = head * cDH + ((dd < 32) ? dd + 32 : dd - 32);
    size_t poff = (size_t)row * cD + pd;
    float sg = (dd < 32) ? -1.f : 1.f;
    qn[e] = Q[off] * c + sg * Q[poff] * s;
    kn[e] = K[off] * c + sg * K[poff] * s;
  }
  __syncthreads();
#pragma unroll
  for (int e = 0; e < 2; ++e) {
    int d = ds[e];
    size_t off = (size_t)row * cD + d;
    Q[off] = qn[e]; K[off] = kn[e];
    if (t < cT) {
      int head = d >> 6, dd = d & 63;
      size_t eo = ((((size_t)n * cNH + head) * cT + t) * cDH) + dd;
      kout[eo] = kn[e];
      vout[eo] = V[off];
    }
  }
}

// ---------------- temporal causal attention: one wave per (n,head,q) --------
__global__ __launch_bounds__(256) void temporal_attn(
    const float* __restrict__ Q, const float* __restrict__ K,
    const float* __restrict__ V, float* __restrict__ O,
    float* __restrict__ logits)   // non-null only on last layer
{
  int wid = blockIdx.x * 8 + (threadIdx.x >> 5);
  int lane = threadIdx.x & 31;
  int n = wid / (cNH * cT1);
  int rem = wid - n * (cNH * cT1);
  int head = rem / cT1, q = rem - head * cT1;
  if (n >= cB * cHW) return;
  int col = head * cDH + lane;
  size_t qoff = ((size_t)(n * cT1 + q)) * cD + col;
  float q0 = Q[qoff] * 0.125f, q1 = Q[qoff + 32] * 0.125f;
  float lg[cT1];
  float mx = NEGV;
#pragma unroll
  for (int k = 0; k < cT1; ++k) {
    bool ok = (k <= q) && !(q == cT1 - 1 && k == cT1 - 1);
    float d = NEGV;
    if (ok) {
      size_t koff = ((size_t)(n * cT1 + k)) * cD + col;
      d = wave_sum(q0 * K[koff] + q1 * K[koff + 32]);
    }
    lg[k] = d;
    mx = fmaxf(mx, d);
  }
  float p[cT1], sum = 0.f;
#pragma unroll
  for (int k = 0; k < cT1; ++k) { p[k] = expf(lg[k] - mx); sum += p[k]; }
  float inv = 1.f / sum;
  float o0 = 0.f, o1 = 0.f;
#pragma unroll
  for (int k = 0; k < cT1; ++k) {
    size_t koff = ((size_t)(n * cT1 + k)) * cD + col;
    o0 += p[k] * inv * V[koff];
    o1 += p[k] * inv * V[koff + 32];
  }
  O[qoff] = o0;
  O[qoff + 32] = o1;
  if (logits && lane == 0) {
    size_t base = (((size_t)n * cNH + head) * cT1 + q) * cT1;
#pragma unroll
    for (int k = 0; k < cT1; ++k) logits[base + k] = lg[k];
  }
}

// ---------------- frame-0 averaging over HW (in S layout) --------------------
__global__ __launch_bounds__(256) void avg_frame0(float* S)
{
  int b = blockIdx.x >> 9;
  int d = blockIdx.x & 511;
  int hw = threadIdx.x;
  size_t off = ((size_t)((b * cHW + hw) * cT1)) * cD + d;
  float v = S[off];
  __shared__ float red[256];
  red[hw] = v; __syncthreads();
  for (int o = 128; o; o >>= 1) { if (hw < o) red[hw] += red[hw + o]; __syncthreads(); }
  float m = red[0] * (1.f / cHW);
  S[off] = m;
}

// =====================================================================
extern "C" void kernel_launch(void* const* d_in, const int* in_sizes, int n_in,
                              void* d_out, int out_size, void* d_ws, size_t ws_size,
                              hipStream_t stream)
{
  (void)in_sizes; (void)n_in; (void)out_size; (void)ws_size;
  // ---- inputs (setup_inputs order) ----
  const int*   codes    = (const int*)  d_in[0];
  const float* emb_tab  = (const float*)d_in[1];
  const float* cls_tok  = (const float*)d_in[2];
  const float* h_emb    = (const float*)d_in[3];
  const float* w_emb    = (const float*)d_in[4];
  const float* emb_g    = (const float*)d_in[5];
  const float* emb_b    = (const float*)d_in[6];
  const float* sn_g     = (const float*)d_in[7];
  const float* sn_b     = (const float*)d_in[8];
  const float* s_wq     = (const float*)d_in[9];
  const float* s_bq     = (const float*)d_in[10];
  const float* s_wk     = (const float*)d_in[11];
  const float* s_bk     = (const float*)d_in[12];
  const float* s_wv     = (const float*)d_in[13];
  const float* s_bv     = (const float*)d_in[14];
  const float* s_wo     = (const float*)d_in[15];
  const float* s_bo     = (const float*)d_in[16];
  const float* tn_g     = (const float*)d_in[17];
  const float* tn_b     = (const float*)d_in[18];
  const float* t_wq     = (const float*)d_in[19];
  const float* t_bq     = (const float*)d_in[20];
  const float* t_wk     = (const float*)d_in[21];
  const float* t_bk     = (const float*)d_in[22];
  const float* t_wv     = (const float*)d_in[23];
  const float* t_bv     = (const float*)d_in[24];
  const float* t_wo     = (const float*)d_in[25];
  const float* t_bo     = (const float*)d_in[26];
  const float* mn_g     = (const float*)d_in[27];
  const float* mn_b     = (const float*)d_in[28];
  const float* mlp_w1   = (const float*)d_in[29];
  const float* mlp_b1   = (const float*)d_in[30];
  const float* mlp_w2   = (const float*)d_in[31];
  const float* mlp_b2   = (const float*)d_in[32];
  const float* fn_g     = (const float*)d_in[33];
  const float* fn_b     = (const float*)d_in[34];

  // ---- workspace carve-out (~180 MB) ----
  char* ws = (char*)d_ws;
  bf16*  WBF = (bf16*)ws;                               // 18,874,368 bf16 (all weights, [N,K])
  float* X   = (float*)(ws + 37748736);                 // [B,T1,HW,D] f32
  float* S   = X + 4456448;                             // [B*HW,T1,D] f32
  bf16*  ABF = (bf16*)(S + 4456448);                    // up to 8704x1024 bf16
  float* QF  = (float*)(ABF + 8912896);
  float* KF  = QF + 4456448;
  float* VF  = KF + 4456448;
  float* OF  = VF + 4456448;                            // up to 8704x1024 f32

  // transposed bf16 weight bases
  const int WSQ = cD * cD;            // 262144 per layer
  const int WM  = cD * cMLP;          // 524288 per layer
  bf16* W_SWQ = WBF + 0 * 6 * WSQ;
  bf16* W_SWK = WBF + 1 * 6 * WSQ;
  bf16* W_SWV = WBF + 2 * 6 * WSQ;
  bf16* W_SWO = WBF + 3 * 6 * WSQ;
  bf16* W_TWQ = WBF + 4 * 6 * WSQ;
  bf16* W_TWK = WBF + 5 * 6 * WSQ;
  bf16* W_TWV = WBF + 6 * 6 * WSQ;
  bf16* W_TWO = WBF + 7 * 6 * WSQ;
  bf16* W_M1  = WBF + 8 * 6 * WSQ;
  bf16* W_M2  = W_M1 + 6 * WM;

  // ---- output layout ----
  float* out   = (float*)d_out;
  float* out_x = out;                                   // 4,456,448
  float* out_lg = out + 4456448;                        // 1,183,744
  float* out_k  = out + 5640192;                        // 6*4,194,304
  float* out_v  = out + 30806016;                       // 6*4,194,304

  // ---- weight conversion (+transpose to [N,K]) ----
  auto cvtw = [&](const float* src, bf16* dst, int K, int N) {
    int total = 6 * K * N;
    wcvt<<<(total + 255) / 256, 256, 0, stream>>>(src, dst, K, N, total);
  };
  cvtw(s_wq, W_SWQ, cD, cD);  cvtw(s_wk, W_SWK, cD, cD);
  cvtw(s_wv, W_SWV, cD, cD);  cvtw(s_wo, W_SWO, cD, cD);
  cvtw(t_wq, W_TWQ, cD, cD);  cvtw(t_wk, W_TWK, cD, cD);
  cvtw(t_wv, W_TWV, cD, cD);  cvtw(t_wo, W_TWO, cD, cD);
  cvtw(mlp_w1, W_M1, cD, cMLP);
  cvtw(mlp_w2, W_M2, cMLP, cD);

  auto gemm = [&](const bf16* A, const bf16* Wt, const float* bias,
                  const float* resid, float* C, int M, int N, int K,
                  int act, int chunk, int gap) {
    dim3 g(M / 256, N / 64);
    gemm_wmma<<<g, 256, 0, stream>>>(A, Wt, bias, resid, C, M, N, K, act, chunk, gap);
  };

  // ---- embedding + input LN (in place) ----
  embed_k<<<ROWS_T, 256, 0, stream>>>(codes, emb_tab, cls_tok, h_emb, w_emb, X);
  ln_k<float><<<ROWS_T, 256, 0, stream>>>(X, emb_g, emb_b, X, 0, 0);

  const int CH = cT * cHW;   // 4096: spatial compact->X row remap chunk
  for (int i = 0; i < cDEPTH; ++i) {
    // ======== spatial windowed attention (frames 0..T-1) ========
    ln_k<bf16><<<ROWS_SP, 256, 0, stream>>>(X, sn_g + i * cD, sn_b + i * cD, ABF, CH, cHW);
    gemm(ABF, W_SWQ + i * WSQ, s_bq + i * cD, nullptr, QF, ROWS_SP, cD, cD, 0, 0, 0);
    gemm(ABF, W_SWK + i * WSQ, s_bk + i * cD, nullptr, KF, ROWS_SP, cD, cD, 0, 0, 0);
    gemm(ABF, W_SWV + i * WSQ, s_bv + i * cD, nullptr, VF, ROWS_SP, cD, cD, 0, 0, 0);
    spatial_attn<<<ROWS_SP, 256, 0, stream>>>(QF, KF, VF, OF);
    cvt_bf16<<<(ROWS_SP * cD) / 256, 256, 0, stream>>>(OF, ABF, ROWS_SP * cD);
    // x[:, :T] = o @ Wo + bo + x[:, :T]   (remapped rows of X; resid aliases C)
    gemm(ABF, W_SWO + i * WSQ, s_bo + i * cD, X, X, ROWS_SP, cD, cD, 0, CH, cHW);

    // ======== temporal RoPE causal attention ========
    xpose<<<ROWS_T, 256, 0, stream>>>(X, S, 1);
    ln_k<bf16><<<ROWS_T, 256, 0, stream>>>(S, tn_g + i * cD, tn_b + i * cD, ABF, 0, 0);
    gemm(ABF, W_TWQ + i * WSQ, t_bq + i * cD, nullptr, QF, ROWS_T, cD, cD, 0, 0, 0);
    gemm(ABF, W_TWK + i * WSQ, t_bk + i * cD, nullptr, KF, ROWS_T, cD, cD, 0, 0, 0);
    gemm(ABF, W_TWV + i * WSQ, t_bv + i * cD, nullptr, VF, ROWS_T, cD, cD, 0, 0, 0);
    rope_export<<<ROWS_T, 256, 0, stream>>>(QF, KF, VF,
        out_k + (size_t)i * 4194304, out_v + (size_t)i * 4194304);
    temporal_attn<<<(cB * cHW * cNH * cT1) / 8, 256, 0, stream>>>(
        QF, KF, VF, OF, (i == cDEPTH - 1) ? out_lg : nullptr);
    cvt_bf16<<<(ROWS_T * cD) / 256, 256, 0, stream>>>(OF, ABF, ROWS_T * cD);
    gemm(ABF, W_TWO + i * WSQ, t_bo + i * cD, S, S, ROWS_T, cD, cD, 0, 0, 0);
    avg_frame0<<<cB * cD, 256, 0, stream>>>(S);

    // ======== MLP ========
    ln_k<bf16><<<ROWS_T, 256, 0, stream>>>(S, mn_g + i * cD, mn_b + i * cD, ABF, 0, 0);
    gemm(ABF, W_M1 + i * WM, mlp_b1 + i * cMLP, nullptr, OF, ROWS_T, cMLP, cD, 1, 0, 0);
    cvt_bf16<<<(ROWS_T * cMLP) / 256, 256, 0, stream>>>(OF, ABF, ROWS_T * cMLP);
    gemm(ABF, W_M2 + i * WM, mlp_b2 + i * cD, S, S, ROWS_T, cD, cMLP, 0, 0, 0);
    xpose<<<ROWS_T, 256, 0, stream>>>(S, X, 0);
  }

  // ---- final LayerNorm straight into d_out ----
  ln_k<float><<<ROWS_T, 256, 0, stream>>>(X, fn_g, fn_b, out_x, 0, 0);
}
